// DUM_3453153706630
// MI455X (gfx1250) — matrix operand order repaired and checked by reading, not compiled, not run on hardware
//
#include <hip/hip_runtime.h>

// ---------------------------------------------------------------------------
// Types for CDNA5 WMMA / TDM
// ---------------------------------------------------------------------------
typedef __attribute__((ext_vector_type(16))) __bf16 v16bf;
typedef __attribute__((ext_vector_type(8)))  __bf16 v8bf;
typedef __attribute__((ext_vector_type(8)))  float  v8f;
typedef __attribute__((ext_vector_type(2)))  float  v2f;
typedef __attribute__((ext_vector_type(4)))  unsigned int v4u;
typedef __attribute__((ext_vector_type(8)))  int    v8i;
typedef __attribute__((ext_vector_type(4)))  int    v4i;

#if defined(__has_builtin)
#if __has_builtin(__builtin_amdgcn_tensor_load_to_lds)
#define HAVE_TDM 1
#endif
#endif
#ifndef HAVE_TDM
#define HAVE_TDM 0
#endif

#define D128 128
#define BPITCH 136            // LDS pitch (bf16) for W^T
#define APITCH 132            // LDS pitch (f32) for A tiles (bank spread)
#define ABUFF  (64 * APITCH)  // floats per A ping-pong buffer (8448)
#define GP 132                // LDS pitch (f32) for Jacobi G/V

// ---------------------------------------------------------------------------
// Degree / norm kernels
// ---------------------------------------------------------------------------
__global__ void deg_init_kernel(float* __restrict__ deg, float* __restrict__ G, int M) {
  int i = blockIdx.x * blockDim.x + threadIdx.x;
  if (i < M) deg[i] = 1.0f;                 // self-loop contributes 1 to every node
  if (i < D128 * D128) G[i] = 0.0f;         // zero Gram accumulator
}

__global__ void deg_edges_kernel(const long long* __restrict__ ei, float* __restrict__ deg, int E) {
  int e = blockIdx.x * blockDim.x + threadIdx.x;
  if (e < E) atomicAdd(&deg[(int)ei[(size_t)E + e]], 1.0f);
}

__global__ void dinv_kernel(const float* __restrict__ deg, float* __restrict__ dinv, int M) {
  int i = blockIdx.x * blockDim.x + threadIdx.x;
  if (i < M) {
    float d = deg[i];
    dinv[i] = d > 0.0f ? rsqrtf(d) : 0.0f;
  }
}

// ---------------------------------------------------------------------------
// Persistent-grid 128-wide GEMM.
//  * W staged into LDS by the Tensor Data Mover (one 2D D# descriptor,
//    TENSORcnt), then converted to transposed bf16 Bs. Fallback: global reads.
//  * A tiles double-buffered with GLOBAL_LOAD_ASYNC_TO_LDS_B128 (ASYNCcnt)
//    while v_wmma_f32_16x16x32_bf16 consumes the current tile.
//   MODE 0: out = acc
//   MODE 1: out = relu(acc + bias)
//   MODE 2: out = (acc + bias) * scores[row] + resid[row, col]
// ---------------------------------------------------------------------------
template <int MODE>
__global__ __launch_bounds__(128) void gemm128_kernel(
    const float* __restrict__ A, const float* __restrict__ W,
    const float* __restrict__ bias, const float* __restrict__ scores,
    const float* __restrict__ resid, float* __restrict__ out,
    int M, int numTiles)
{
  extern __shared__ float smem[];
  float*  As = smem;                           // [2][64][APITCH] f32 (16896 floats)
  __bf16* Bs = (__bf16*)(smem + 2 * ABUFF);    // [128][BPITCH]   bf16 W^T

  const int tid  = threadIdx.x;
  const int wave = tid >> 5;
  const int lane = tid & 31;
  const int l16  = lane & 15;
  const int lhi  = lane >> 4;

#if HAVE_TDM
  // ---- Stage W (f32 row-major, 64KB) into the As region via TDM ----
  // D# group0: count=1, lds_addr, global_addr, type=2.
  // D# group1: data_size=4B, tensor 128x128, tile 128x128, dim0 stride 128.
  if (wave == 0) {
    unsigned ldsW = (unsigned)(size_t)(const void*)As;
    unsigned long long ga = (unsigned long long)(size_t)(const void*)W;
    v4u g0;
    g0[0] = 1u;                                        // count=1 (valid descriptor)
    g0[1] = ldsW;                                      // lds_addr
    g0[2] = (unsigned)(ga & 0xffffffffu);              // global_addr[31:0]
    g0[3] = (unsigned)((ga >> 32) & 0x01ffffffu) | 0x80000000u;  // addr[56:32] | type=2
    v8i g1 = { (int)0x00020000,   // data_size=2 (4B), no pad/iterate/multicast
               (int)0x00800000,   // tensor_dim0=128 (low 16 bits at bit 48)
               (int)0x00800000,   // tensor_dim1=128 (low 16 bits at bit 80)
               (int)0x00800000,   // tile_dim0=128   (bits 127:112)
               128,               // tile_dim1=128   (bits 143:128)
               128,               // tensor_dim0_stride=128 (bits 191:160)
               0, 0 };
    v4i g2 = {}, g3 = {};
#if __clang_major__ >= 23
    v8i g4 = {};
    __builtin_amdgcn_tensor_load_to_lds(g0, g1, g2, g3, g4, 0);
#else
    __builtin_amdgcn_tensor_load_to_lds(g0, g1, g2, g3, 0);
#endif
  }
  __builtin_amdgcn_s_wait_tensorcnt((short)0);
  __syncthreads();                       // W tile visible to all waves
  {
    const float* Wstage = As;
    for (int idx = tid; idx < D128 * D128; idx += 128) {
      int k = idx >> 7, n = idx & 127;
      Bs[n * BPITCH + k] = (__bf16)Wstage[idx];
    }
  }
  __syncthreads();                       // conversion done before As is reused
#else
  // ---- Fallback: coalesced global reads of W, transposed bf16 store ----
  for (int idx = tid; idx < D128 * D128; idx += 128) {
    int k = idx >> 7, n = idx & 127;
    Bs[n * BPITCH + k] = (__bf16)W[idx];
  }
  // (first in-loop barrier covers Bs visibility)
#endif

  // Async-stage one 64-row A tile (16 x b128 per thread, per-lane LDS addrs).
  auto issueTile = [&](int mt, int buf) {
#pragma unroll
    for (int i = 0; i < 16; ++i) {
      int id   = tid + 128 * i;     // 0..2047 chunks of 16B
      int row  = id >> 5;           // 0..63
      int c    = id & 31;           // 16B chunk within the 512B row
      int grow = mt * 64 + row; if (grow >= M) grow = M - 1;   // clamp (stores masked later)
      const float* gp = A + (size_t)grow * D128 + c * 4;
      unsigned ldsb = (unsigned)(size_t)(const void*)(As + buf * ABUFF + row * APITCH + c * 4);
      asm volatile("global_load_async_to_lds_b128 %0, %1, off"
                   :: "v"(ldsb), "v"(gp) : "memory");
    }
  };

  int mt  = blockIdx.x;
  int cur = 0;
  if (mt < numTiles) issueTile(mt, 0);

  for (; mt < numTiles; mt += gridDim.x) {
    const int nxt = mt + gridDim.x;
    if (nxt < numTiles) {
      issueTile(nxt, cur ^ 1);
      asm volatile("s_wait_asynccnt 0x10" ::: "memory");  // oldest 16 (current tile) done
    } else {
      asm volatile("s_wait_asynccnt 0x0" ::: "memory");   // drain
    }
    __syncthreads();   // current buffer visible to all waves

    const float* Arow = As + cur * ABUFF + (wave * 16 + l16) * APITCH;

    v8f acc[8] = {};
#pragma unroll
    for (int ks = 0; ks < 4; ++ks) {
      // A fragment: lane holds row l16; VGPR0..3 = K[k0..k0+7], VGPR4..7 = K[k0+16..+23]
      v16bf af;
      const int k0 = ks * 32 + lhi * 8;
#pragma unroll
      for (int j = 0; j < 8; ++j) {
        af[j]     = (__bf16)Arow[k0 + j];
        af[8 + j] = (__bf16)Arow[k0 + 16 + j];
      }
#pragma unroll
      for (int n = 0; n < 8; ++n) {
        const v8bf* bp = (const v8bf*)&Bs[(n * 16 + l16) * BPITCH + ks * 32 + lhi * 8];
        v8bf b0 = bp[0], b1 = bp[2];
        v16bf bf;
#pragma unroll
        for (int j = 0; j < 8; ++j) { bf[j] = b0[j]; bf[8 + j] = b1[j]; }
        acc[n] = __builtin_amdgcn_wmma_f32_16x16x32_bf16(
            false, af, false, bf, (short)0, acc[n], false, false);
      }
    }

    // Epilogue: branch once per wave-tile instead of per element.
    const int wbase = mt * 64 + wave * 16;
    if (wbase + 15 < M) {
#pragma unroll
      for (int n = 0; n < 8; ++n) {
        const int col = n * 16 + l16;
#pragma unroll
        for (int r = 0; r < 8; ++r) {
          const int row = wbase + lhi * 8 + r;
          float v = acc[n][r];
          if (MODE == 1) { v += bias[col]; v = v > 0.0f ? v : 0.0f; }
          if (MODE == 2) { v += bias[col]; v = v * scores[row] + resid[(size_t)row * D128 + col]; }
          out[(size_t)row * D128 + col] = v;
        }
      }
    } else {
#pragma unroll
      for (int n = 0; n < 8; ++n) {
        const int col = n * 16 + l16;
#pragma unroll
        for (int r = 0; r < 8; ++r) {
          const int row = wbase + lhi * 8 + r;
          if (row < M) {
            float v = acc[n][r];
            if (MODE == 1) { v += bias[col]; v = v > 0.0f ? v : 0.0f; }
            if (MODE == 2) { v += bias[col]; v = v * scores[row] + resid[(size_t)row * D128 + col]; }
            out[(size_t)row * D128 + col] = v;
          }
        }
      }
    }
    __syncthreads();   // everyone done reading buf `cur` before it is re-staged
    cur ^= 1;
  }
}

// ---------------------------------------------------------------------------
// GCN aggregation
// ---------------------------------------------------------------------------
__global__ void agg_init_kernel(const float* __restrict__ h, const float* __restrict__ bg,
                                const float* __restrict__ dinv, float* __restrict__ hagg, int M) {
  int idx = blockIdx.x * blockDim.x + threadIdx.x;
  if (idx < M * D128) {
    int i = idx >> 7, j = idx & 127;
    float di = dinv[i];
    hagg[idx] = bg[j] + di * di * h[idx];   // bias + self-loop term
  }
}

__global__ void agg_edges_kernel(const long long* __restrict__ ei, const float* __restrict__ dinv,
                                 const float* __restrict__ h, float* __restrict__ hagg, int E) {
  int idx = blockIdx.x * blockDim.x + threadIdx.x;   // 32 threads per edge
  int e = idx >> 5;
  int lane = idx & 31;
  if (e >= E) return;
  int s = (int)ei[e];
  int d = (int)ei[(size_t)E + e];
  float nrm = dinv[s] * dinv[d];
  float4 hv = ((const float4*)(h + (size_t)s * D128))[lane];
  float* dp = hagg + (size_t)d * D128 + lane * 4;
  atomicAdd(dp + 0, nrm * hv.x);
  atomicAdd(dp + 1, nrm * hv.y);
  atomicAdd(dp + 2, nrm * hv.z);
  atomicAdd(dp + 3, nrm * hv.w);
}

// One wave32 per row: dot(h_agg[i], w_score) -> sigmoid
__global__ __launch_bounds__(128) void score_kernel(const float* __restrict__ hagg,
                                                    const float* __restrict__ wsc,
                                                    const float* __restrict__ bsc,
                                                    float* __restrict__ scores, int M) {
  int row  = blockIdx.x * 4 + (threadIdx.x >> 5);
  int lane = threadIdx.x & 31;
  if (row >= M) return;
  float4 a = ((const float4*)(hagg + (size_t)row * D128))[lane];
  float4 b = ((const float4*)wsc)[lane];
  float s = a.x * b.x + a.y * b.y + a.z * b.z + a.w * b.w;
#pragma unroll
  for (int off = 16; off; off >>= 1) s += __shfl_xor(s, off, 32);
  if (lane == 0) scores[row] = 1.0f / (1.0f + __expf(-(s + bsc[0])));
}

__global__ void gate_kernel(const long long* __restrict__ ei, const float* __restrict__ ew,
                            const float* __restrict__ scores, const float* __restrict__ alpha,
                            const float* __restrict__ beta, float* __restrict__ out_e, int E) {
  int e = blockIdx.x * blockDim.x + threadIdx.x;
  if (e >= E) return;
  int s = (int)ei[e];
  int d = (int)ei[(size_t)E + e];
  float g = 1.0f / (1.0f + __expf(-(alpha[0] * (scores[s] + scores[d]) + beta[0])));
  out_e[e] = ew[e] * g;
}

// ---------------------------------------------------------------------------
// Gram: G = x^T x (128x128) in fp32 WMMA 16x16x4, split-N atomic reduce.
// ---------------------------------------------------------------------------
__global__ __launch_bounds__(128) void gram_kernel(const float* __restrict__ X,
                                                   float* __restrict__ G,
                                                   int M, int rowsPerBlock) {
  const int tid  = threadIdx.x;
  const int wave = tid >> 5;
  const int lane = tid & 31;
  const int l16  = lane & 15;
  const int lhi  = lane >> 4;
  const int n0 = blockIdx.x * rowsPerBlock;
  int n1 = n0 + rowsPerBlock; if (n1 > M) n1 = M;
  const int cA0 = wave * 32;       // this wave's A column base (rows of G)
  const int cA1 = wave * 32 + 16;

  v8f acc0[8] = {}, acc1[8] = {};

  int nb = n0;
  const int n1full = n0 + ((n1 - n0) & ~3);
  for (; nb < n1full; nb += 4) {
    const float* r0 = X + (size_t)(nb + lhi * 2) * D128 + l16;
    const float* r1 = r0 + D128;
    v2f f[8], a0, a1;
#pragma unroll
    for (int c = 0; c < 8; ++c) { f[c][0] = r0[c * 16]; f[c][1] = r1[c * 16]; }
    a0[0] = r0[cA0]; a0[1] = r1[cA0];
    a1[0] = r0[cA1]; a1[1] = r1[cA1];
#pragma unroll
    for (int c = 0; c < 8; ++c) {
      acc0[c] = __builtin_amdgcn_wmma_f32_16x16x4_f32(false, a0, false, f[c], (short)0, acc0[c], false, false);
      acc1[c] = __builtin_amdgcn_wmma_f32_16x16x4_f32(false, a1, false, f[c], (short)0, acc1[c], false, false);
    }
  }
  if (nb < n1) {  // tail: branchless clamp + zero-mask
    v2f f[8], a0, a1;
    int   kr[2];
    float mk[2];
#pragma unroll
    for (int j = 0; j < 2; ++j) {
      int k = nb + lhi * 2 + j;
      mk[j] = (k < n1) ? 1.0f : 0.0f;
      kr[j] = (k < M) ? k : (M - 1);
    }
#pragma unroll
    for (int c = 0; c < 8; ++c) {
#pragma unroll
      for (int j = 0; j < 2; ++j)
        f[c][j] = X[(size_t)kr[j] * D128 + c * 16 + l16] * mk[j];
    }
#pragma unroll
    for (int j = 0; j < 2; ++j) {
      a0[j] = X[(size_t)kr[j] * D128 + cA0 + l16] * mk[j];
      a1[j] = X[(size_t)kr[j] * D128 + cA1 + l16] * mk[j];
    }
#pragma unroll
    for (int c = 0; c < 8; ++c) {
      acc0[c] = __builtin_amdgcn_wmma_f32_16x16x4_f32(false, a0, false, f[c], (short)0, acc0[c], false, false);
      acc1[c] = __builtin_amdgcn_wmma_f32_16x16x4_f32(false, a1, false, f[c], (short)0, acc1[c], false, false);
    }
  }

#pragma unroll
  for (int c = 0; c < 8; ++c) {
#pragma unroll
    for (int r = 0; r < 8; ++r) {
      const int col  = c * 16 + l16;
      const int row0 = cA0 + lhi * 8 + r;
      const int row1 = cA1 + lhi * 8 + r;
      atomicAdd(&G[row0 * D128 + col], acc0[c][r]);
      atomicAdd(&G[row1 * D128 + col], acc1[c][r]);
    }
  }
}

// ---------------------------------------------------------------------------
// Parallel cyclic Jacobi eigensolve of G (symmetric 128x128), single block.
// G and V live in dynamic LDS (132 KB of the 320 KB WGP budget).
// Emits Mf = 0.5 * (I + V_k V_k^T), fusing the 0.5x + 0.5x_den blend.
// ---------------------------------------------------------------------------
__global__ __launch_bounds__(128) void jacobi_kernel(const float* __restrict__ Gin,
                                                     float* __restrict__ Mf) {
  extern __shared__ float lds[];
  float* Gs = lds;               // [128][GP]
  float* Vs = lds + D128 * GP;   // [128][GP]

  __shared__ int   pa[128];
  __shared__ float cs[64], sn[64];
  __shared__ int   pp[64], qq[64];
  __shared__ float Sv[128];
  __shared__ int   ord[128];
  __shared__ int   kkeep;

  const int t = threadIdx.x;

  for (int idx = t; idx < D128 * D128; idx += 128) {
    int i = idx >> 7, j = idx & 127;
    Gs[i * GP + j] = Gin[idx];
    Vs[i * GP + j] = (i == j) ? 1.0f : 0.0f;
  }
  pa[t] = t;
  __syncthreads();

  for (int sweep = 0; sweep < 10; ++sweep) {
    for (int round = 0; round < 127; ++round) {
      if (t < 64) {
        int p = pa[t], q = pa[127 - t];
        if (p > q) { int z = p; p = q; q = z; }
        float app = Gs[p * GP + p], aqq = Gs[q * GP + q], apq = Gs[p * GP + q];
        float c = 1.0f, s = 0.0f;
        if (fabsf(apq) > 1e-12f) {
          float tau = (aqq - app) / (2.0f * apq);
          float tt  = (tau >= 0.0f ? 1.0f : -1.0f) / (fabsf(tau) + sqrtf(1.0f + tau * tau));
          c = rsqrtf(1.0f + tt * tt);
          s = tt * c;
        }
        pp[t] = p; qq[t] = q; cs[t] = c; sn[t] = s;
      }
      __syncthreads();
      // Left multiply J^T: rows p,q. Thread t owns column t; pair rows disjoint.
#pragma unroll 4
      for (int m = 0; m < 64; ++m) {
        float c = cs[m], s = sn[m];
        int p = pp[m], q = qq[m];
        float gp = Gs[p * GP + t], gq = Gs[q * GP + t];
        Gs[p * GP + t] = c * gp - s * gq;
        Gs[q * GP + t] = s * gp + c * gq;
      }
      __syncthreads();
      // Right multiply J: columns p,q of G and V. Thread t owns row t.
#pragma unroll 4
      for (int m = 0; m < 64; ++m) {
        float c = cs[m], s = sn[m];
        int p = pp[m], q = qq[m];
        float gp = Gs[t * GP + p], gq = Gs[t * GP + q];
        Gs[t * GP + p] = c * gp - s * gq;
        Gs[t * GP + q] = s * gp + c * gq;
        float vp = Vs[t * GP + p], vq = Vs[t * GP + q];
        Vs[t * GP + p] = c * vp - s * vq;
        Vs[t * GP + q] = s * vp + c * vq;
      }
      __syncthreads();
      if (t == 0) {  // round-robin tournament rotation (pa[0] fixed)
        int last = pa[127];
        for (int i = 127; i > 1; --i) pa[i] = pa[i - 1];
        pa[1] = last;
      }
      __syncthreads();
    }
  }

  // Singular values S = sqrt(eig), sort descending, derive data-dependent k.
  Sv[t] = sqrtf(fmaxf(Gs[t * GP + t], 0.0f));
  ord[t] = t;
  __syncthreads();
  if (t == 0) {
    for (int i = 1; i < 128; ++i) {
      float sv = Sv[i]; int oi = ord[i]; int j = i - 1;
      while (j >= 0 && Sv[j] < sv) { Sv[j + 1] = Sv[j]; ord[j + 1] = ord[j]; --j; }
      Sv[j + 1] = sv; ord[j + 1] = oi;
    }
    int cnt = 0;
    for (int i = 0; i < 127; ++i)
      if (Sv[i + 1] - Sv[i] > 0.1f) ++cnt;   // (1 - THRESH)
    kkeep = cnt + 1;
  }
  __syncthreads();
  const int k = kkeep;

  // Mf = 0.5 * (I + V_k V_k^T)
  for (int idx = t; idx < D128 * D128; idx += 128) {
    int i = idx >> 7, j = idx & 127;
    float s = (i == j) ? 1.0f : 0.0f;
    for (int r = 0; r < k; ++r) {
      int c = ord[r];
      s += Vs[i * GP + c] * Vs[j * GP + c];
    }
    Mf[idx] = 0.5f * s;
  }
}

// ---------------------------------------------------------------------------
// Host launcher
// ---------------------------------------------------------------------------
extern "C" void kernel_launch(void* const* d_in, const int* in_sizes, int n_in,
                              void* d_out, int out_size, void* d_ws, size_t ws_size,
                              hipStream_t stream) {
  const float*     x       = (const float*)d_in[0];
  const long long* ei      = (const long long*)d_in[1];
  const float*     ew      = (const float*)d_in[2];
  const float*     w_gcn   = (const float*)d_in[3];
  const float*     b_gcn   = (const float*)d_in[4];
  const float*     w_score = (const float*)d_in[5];
  const float*     b_score = (const float*)d_in[6];
  const float*     alpha   = (const float*)d_in[7];
  const float*     beta    = (const float*)d_in[8];
  const float*     w_e1    = (const float*)d_in[9];
  const float*     b_e1    = (const float*)d_in[10];
  const float*     w_e2    = (const float*)d_in[11];
  const float*     b_e2    = (const float*)d_in[12];

  const int M = in_sizes[0] / D128;   // 50000
  const int E = in_sizes[2];          // 800000

  // Workspace layout (buffers reused along the pipeline timeline):
  float* ws     = (float*)d_ws;
  float* deg    = ws;                          // [M]
  float* dinv   = deg + M;                     // [M]
  float* scores = dinv + M;                    // [M]
  float* bufA   = scores + M;                  // [M*128]  h, then xf
  float* bufB   = bufA + (size_t)M * D128;     // [M*128]  h_agg, then relu tmp
  float* G      = bufB + (size_t)M * D128;     // [128*128]
  float* Mf     = G + D128 * D128;             // [128*128]

  float* out_x = (float*)d_out;                // x_enh [M*128]
  float* out_e = out_x + (size_t)M * D128;     // ew_out [E]

  const dim3 b256(256), b128(128);
  const int initN = (M > D128 * D128) ? M : D128 * D128;

  const int numTiles = (M + 63) / 64;
  const int gemmGrid = numTiles < 256 ? numTiles : 256;          // persistent-ish grid
  const size_t gemmLds = (size_t)2 * ABUFF * sizeof(float)       // A ping-pong (f32)
                       + (size_t)D128 * BPITCH * sizeof(__bf16); // W^T (bf16)  ~= 100 KB

  // 1) degree (self-loops) + zero Gram
  deg_init_kernel<<<dim3((initN + 255) / 256), b256, 0, stream>>>(deg, G, M);
  deg_edges_kernel<<<dim3((E + 255) / 256), b256, 0, stream>>>(ei, deg, E);
  dinv_kernel<<<dim3((M + 255) / 256), b256, 0, stream>>>(deg, dinv, M);

  // 2) h = x @ w_gcn   (TDM W-stage + async double-buffered A + bf16 WMMA)
  gemm128_kernel<0><<<dim3(gemmGrid), b128, gemmLds, stream>>>(
      x, w_gcn, nullptr, nullptr, nullptr, bufA, M, numTiles);

  // 3) GCN aggregation (bias + self-loop init, then edge scatter; L2-resident)
  agg_init_kernel<<<dim3((M * D128 + 255) / 256), b256, 0, stream>>>(bufA, b_gcn, dinv, bufB, M);
  agg_edges_kernel<<<dim3(((size_t)E * 32 + 255) / 256), b256, 0, stream>>>(ei, dinv, bufA, bufB, E);

  // 4) scores = sigmoid(h_agg @ w_score + b_score)
  score_kernel<<<dim3((M + 3) / 4), b128, 0, stream>>>(bufB, w_score, b_score, scores, M);

  // 5) gated edge weights -> second output
  gate_kernel<<<dim3((E + 255) / 256), b256, 0, stream>>>(ei, ew, scores, alpha, beta, out_e, E);

  // 6) Gram G = x^T x  (fp32 WMMA, split-N atomic reduce)
  const int rowsPerBlock = 512;
  gram_kernel<<<dim3((M + rowsPerBlock - 1) / rowsPerBlock), b128, 0, stream>>>(x, G, M, rowsPerBlock);

  // 7) Jacobi eigensolve + fused filter operator Mf = 0.5(I + V_k V_k^T)
  const size_t jacobiLds = (size_t)2 * D128 * GP * sizeof(float);  // 132 KB of 320 KB/WGP
  jacobi_kernel<<<dim3(1), b128, jacobiLds, stream>>>(G, Mf);

  // 8) x_fused = x @ Mf   (into bufA; h is dead)
  gemm128_kernel<0><<<dim3(gemmGrid), b128, gemmLds, stream>>>(
      x, Mf, nullptr, nullptr, nullptr, bufA, M, numTiles);

  // 9) tmp = relu(x_fused @ w_e1 + b_e1)   (into bufB; h_agg is dead)
  gemm128_kernel<1><<<dim3(gemmGrid), b128, gemmLds, stream>>>(
      bufA, w_e1, b_e1, nullptr, nullptr, bufB, M, numTiles);

  // 10) x_enh = (tmp @ w_e2 + b_e2) * scores + x_fused
  gemm128_kernel<2><<<dim3(gemmGrid), b128, gemmLds, stream>>>(
      bufB, w_e2, b_e2, scores, bufA, out_x, M, numTiles);
}
